// MultiHeadAttention_14130442404267
// MI455X (gfx1250) — matrix operand order repaired and checked by reading, not compiled, not run on hardware
//
#include <hip/hip_runtime.h>
#include <hip/hip_bf16.h>

#define BATCH  4
#define SEQ    2048
#define HIDDEN 1024
#define HEADS  16
#define HD     64          // head dim
#define BM 128
#define BN 128
#define BK 32

typedef __bf16 bf16_t;
typedef __attribute__((ext_vector_type(16))) __bf16 v16bf;
typedef __attribute__((ext_vector_type(8)))  __bf16 v8bf;
typedef __attribute__((ext_vector_type(4)))  __bf16 v4bf;
typedef __attribute__((ext_vector_type(8)))  float  v8f;
typedef unsigned int u32x4 __attribute__((ext_vector_type(4)));
typedef int          i32x8 __attribute__((ext_vector_type(8)));
typedef int          i32x4 __attribute__((ext_vector_type(4)));

#if __has_builtin(__builtin_amdgcn_tensor_load_to_lds)
#define HAS_TDM 1
#else
#define HAS_TDM 0
#endif

static __device__ inline v16bf cat8(v8bf lo, v8bf hi) {
  return __builtin_shufflevector(lo, hi, 0,1,2,3,4,5,6,7,8,9,10,11,12,13,14,15);
}
static __device__ inline v8f wmma_bf16(v16bf a, v16bf b, v8f c) {
  return __builtin_amdgcn_wmma_f32_16x16x32_bf16(false, a, false, b, (short)0, c, false, false);
}
static __device__ inline v8f v8f_zero() {
  v8f z = {0.f,0.f,0.f,0.f,0.f,0.f,0.f,0.f};
  return z;
}
static __device__ inline unsigned lds_off(const void* p) {
  // LDS aperture: generic addr low 32 bits are the wave-relative LDS byte offset
  return (unsigned)(uintptr_t)p;
}

#if HAS_TDM
// 2D tile DMA: global (row-major, stride0 elements of 2B) -> contiguous LDS.
// Dims/strides in data_size (=2B) units.  Built per CDNA5 ISA D# layout (CDNA5 ISA).
static __device__ inline void tdm_load_2d(unsigned ldsaddr, const void* gptr,
                                          unsigned tile0, unsigned tile1,
                                          unsigned tdim0, unsigned tdim1,
                                          unsigned stride0) {
  const unsigned long long ga = (unsigned long long)(uintptr_t)gptr;
  u32x4 g0;
  g0[0] = 1u;                                            // count=1, user descriptor
  g0[1] = ldsaddr;                                       // lds_addr [63:32]
  g0[2] = (unsigned)ga;                                  // global_addr [95:64]
  g0[3] = (unsigned)((ga >> 32) & 0x01FFFFFFu) | (2u << 30); // addr[56:32], type=2
  i32x8 g1;
  g1[0] = 0x00010000;                                    // data_size=1 (2 bytes)
  g1[1] = (int)((tdim0 & 0xFFFFu) << 16);                // tensor_dim0[15:0] @ bits 63:48
  g1[2] = (int)(((tdim0 >> 16) & 0xFFFFu) | ((tdim1 & 0xFFFFu) << 16)); // dim0 hi | dim1 lo
  g1[3] = (int)(((tdim1 >> 16) & 0xFFFFu) | (tile0 << 16));             // dim1 hi | tile_dim0
  g1[4] = (int)tile1;                                    // tile_dim1 (tile_dim2 = 0)
  g1[5] = (int)stride0;                                  // tensor_dim0_stride[31:0]
  g1[6] = 0;                                             // stride0 hi | stride1 lo
  g1[7] = 0;
  i32x4 gz = {0, 0, 0, 0};                               // groups 2/3 unused (2D)
#if defined(__clang_major__) && __clang_major__ >= 23
  i32x8 gz8 = {0,0,0,0,0,0,0,0};
  __builtin_amdgcn_tensor_load_to_lds(g0, g1, gz, gz, gz8, 0);
#else
  __builtin_amdgcn_tensor_load_to_lds(g0, g1, gz, gz, 0);
#endif
}
#endif

// ---------------------------------------------------------------- convert
__global__ __launch_bounds__(256)
void f32_to_bf16_kernel(const float* __restrict__ in, bf16_t* __restrict__ out, int n) {
  int i = (blockIdx.x * blockDim.x + threadIdx.x) * 4;
  if (i >= n) return;
  float4 v = *(const float4*)(in + i);
  v4bf o;
  o[0] = (__bf16)v.x; o[1] = (__bf16)v.y; o[2] = (__bf16)v.z; o[3] = (__bf16)v.w;
  *(v4bf*)(out + i) = o;
}

// ---------------------------------------------------------------- GEMM
// C[m,n] = sum_k A[m,k] * W[n,k]   (torch Linear: x @ W^T), W row-major == B col-major.
// TDM double-buffers the 128x32 A/B tiles in LDS; 8 waves x (4x2) WMMA register tiles.
// MODE 0: bf16 [M,N] | 1: bf16 split-heads [B,H,S,D] | 2: bf16 V^T [B,H,D,S] | 3: f32 [M,N]
template<int MODE>
__global__ __launch_bounds__(256)
void gemm_bf16_kernel(const bf16_t* __restrict__ A, const bf16_t* __restrict__ W,
                      const float* __restrict__ bias, void* __restrict__ outp,
                      int M, int N, int K, float scale) {
  __shared__ bf16_t As[2][BM * BK];
  __shared__ bf16_t Bs[2][BN * BK];

  const int m0   = blockIdx.y * BM;
  const int n0   = blockIdx.x * BN;
  const int t    = threadIdx.x;
  const int wid  = t >> 5;
  const int lane = t & 31;
  const int l16  = lane & 15;
  const int lhi  = lane >> 4;
  const int wm   = (wid & 1) * 64;   // 2 wave rows  -> 128 M
  const int wn   = (wid >> 1) * 32;  // 4 wave cols  -> 128 N

  v8f c[4][2];
  for (int i = 0; i < 4; ++i)
    for (int j = 0; j < 2; ++j) c[i][j] = v8f_zero();

  auto stage = [&](int k0, int buf) {
#if HAS_TDM
    if (wid == 0) {
      tdm_load_2d(lds_off(&As[buf][0]), A + (size_t)m0 * K + k0,
                  BK, BM, (unsigned)(K - k0), (unsigned)(M - m0), (unsigned)K);
      tdm_load_2d(lds_off(&Bs[buf][0]), W + (size_t)n0 * K + k0,
                  BK, BN, (unsigned)(K - k0), (unsigned)(N - n0), (unsigned)K);
    }
#else
    const int lrow = t >> 1;
    const int lcol = (t & 1) << 4;
    const bf16_t* ga = A + (size_t)(m0 + lrow) * K + k0 + lcol;
    const bf16_t* gb = W + (size_t)(n0 + lrow) * K + k0 + lcol;
    *(v8bf*)&As[buf][lrow * BK + lcol]     = *(const v8bf*)ga;
    *(v8bf*)&As[buf][lrow * BK + lcol + 8] = *(const v8bf*)(ga + 8);
    *(v8bf*)&Bs[buf][lrow * BK + lcol]     = *(const v8bf*)gb;
    *(v8bf*)&Bs[buf][lrow * BK + lcol + 8] = *(const v8bf*)(gb + 8);
#endif
  };
  auto publish = [&]() {
#if HAS_TDM
    if (wid == 0) __builtin_amdgcn_s_wait_tensorcnt(0);  // DMA done before barrier
#endif
    __syncthreads();
  };

  const int nsteps = K / BK;
  stage(0, 0);
  publish();
  for (int step = 0; step < nsteps; ++step) {
    const int cur = step & 1;
    if (step + 1 < nsteps) stage((step + 1) * BK, cur ^ 1);  // overlap DMA w/ compute

    v16bf af[4], bfr[2];
    for (int i = 0; i < 4; ++i) {
      const bf16_t* p = &As[cur][(wm + i * 16 + l16) * BK];
      af[i] = cat8(*(const v8bf*)(p + lhi * 8), *(const v8bf*)(p + 16 + lhi * 8));
    }
    for (int j = 0; j < 2; ++j) {
      const bf16_t* p = &Bs[cur][(wn + j * 16 + l16) * BK + lhi * 8];
      bfr[j] = cat8(*(const v8bf*)p, *(const v8bf*)(p + 16));
    }
    for (int i = 0; i < 4; ++i)
      for (int j = 0; j < 2; ++j)
        c[i][j] = wmma_bf16(af[i], bfr[j], c[i][j]);

    __syncthreads();                       // everyone done reading `cur`
    if (step + 1 < nsteps) publish();      // next buffer visible to all waves
  }

  // epilogue: C/D layout -> VGPR r: lanes 0-15 row r, lanes 16-31 row r+8; col = lane%16
  const int radd = (lane < 16) ? 0 : 8;
  for (int i = 0; i < 4; ++i)
    for (int j = 0; j < 2; ++j) {
      const int nl = n0 + wn + j * 16 + l16;
      const float bvv = bias[nl];
      for (int r = 0; r < 8; ++r) {
        const int ml = m0 + wm + i * 16 + r + radd;
        const float v = (c[i][j][r] + bvv) * scale;
        if (MODE == 3) {
          ((float*)outp)[(size_t)ml * N + nl] = v;
        } else {
          const bf16_t o = (bf16_t)v;
          if (MODE == 0) {
            ((bf16_t*)outp)[(size_t)ml * N + nl] = o;
          } else {
            const int bb = ml / SEQ, s = ml % SEQ;
            const int hh = nl / HD,  d = nl % HD;
            if (MODE == 1)
              ((bf16_t*)outp)[(((size_t)bb * HEADS + hh) * SEQ + s) * HD + d] = o;
            else
              ((bf16_t*)outp)[(((size_t)bb * HEADS + hh) * HD + d) * SEQ + s] = o;
          }
        }
      }
    }
}

// ---------------------------------------------------------------- flash attention
// grid (S/128, H, B); 8 waves/block, each wave owns 16 query rows.
// K/V 32-key chunks are TDM-DMA'd into double-buffered LDS shared by all 8 waves.
__global__ __launch_bounds__(256)
void attention_kernel(const bf16_t* __restrict__ Qh, const bf16_t* __restrict__ Kh,
                      const bf16_t* __restrict__ Vt, const float* __restrict__ masks,
                      bf16_t* __restrict__ attn) {
  __shared__ bf16_t Ks[2][32 * HD];    // [key][d]   4 KB x2
  __shared__ bf16_t Vs[2][HD * 32];    // [d][key]   4 KB x2
  __shared__ bf16_t Pst[8][16 * 32];   // per-wave P staging (C-layout -> A-layout)

  const int b    = blockIdx.z;
  const int h    = blockIdx.y;
  const int t    = threadIdx.x;
  const int wid  = t >> 5;
  const int lane = t & 31;
  const int l16  = lane & 15;
  const int lhi  = lane >> 4;
  const int qbase = blockIdx.x * 128 + wid * 16;

  const size_t headRow = ((size_t)b * HEADS + h) * SEQ;
  const bf16_t* Qp = Qh + (headRow + qbase) * HD;
  const bf16_t* Kp = Kh + headRow * HD;
  const bf16_t* Vp = Vt + ((size_t)b * HEADS + h) * (size_t)HD * SEQ;
  const float*  mk = masks + (size_t)b * SEQ;

  auto stage = [&](int kt, int buf) {
#if HAS_TDM
    if (wid == 0) {
      tdm_load_2d(lds_off(&Ks[buf][0]), Kp + (size_t)kt * HD,
                  HD, 32, HD, (unsigned)(SEQ - kt), HD);
      tdm_load_2d(lds_off(&Vs[buf][0]), Vp + kt,
                  32, HD, (unsigned)(SEQ - kt), HD, SEQ);
    }
#else
    const int idx = t * 8;             // 2048 elems each, 8 per thread
    *(v8bf*)&Ks[buf][idx] = *(const v8bf*)(Kp + (size_t)(kt + (idx >> 6)) * HD + (idx & 63));
    *(v8bf*)&Vs[buf][idx] = *(const v8bf*)(Vp + (size_t)(idx >> 5) * SEQ + kt + (idx & 31));
#endif
  };
  auto publish = [&]() {
#if HAS_TDM
    if (wid == 0) __builtin_amdgcn_s_wait_tensorcnt(0);
#endif
    __syncthreads();
  };

  // Q A-frags: d = 0..31 and 32..63 (Q pre-scaled by D^-0.5 in projection)
  v16bf aq[2];
  for (int tt = 0; tt < 2; ++tt) {
    const bf16_t* p = Qp + l16 * HD + tt * 32 + lhi * 8;
    aq[tt] = cat8(*(const v8bf*)p, *(const v8bf*)(p + 16));
  }

  float mrow[8], srow[8];
  v8f acc[4];
  for (int r = 0; r < 8; ++r) { mrow[r] = -1.0e30f; srow[r] = 0.f; }
  for (int j = 0; j < 4; ++j) acc[j] = v8f_zero();

  const int nsteps = SEQ / 32;
  stage(0, 0);
  publish();
  for (int step = 0; step < nsteps; ++step) {
    const int kt  = step * 32;
    const int cur = step & 1;
    if (step + 1 < nsteps) stage(kt + 32, cur ^ 1);   // DMA next chunk during compute

    // ---- logits: two 16-key subtiles, contraction over d=64 (2 WMMA each)
    v8f lg[2];
    for (int sub = 0; sub < 2; ++sub) {
      v8f cl = v8f_zero();
      for (int tt = 0; tt < 2; ++tt) {
        const bf16_t* p = &Ks[cur][(sub * 16 + l16) * HD + tt * 32 + lhi * 8];
        v16bf bb = cat8(*(const v8bf*)p, *(const v8bf*)(p + 16));
        cl = wmma_bf16(aq[tt], bb, cl);
      }
      const float mb = (1.0f - mk[kt + sub * 16 + l16]) * -1.0e9f;
      for (int r = 0; r < 8; ++r) cl[r] += mb;
      lg[sub] = cl;
    }

    // ---- online softmax (row-wise over keys; rows live across 16-lane groups)
    v8f p0, p1;
    for (int r = 0; r < 8; ++r) {
      float mx = fmaxf(lg[0][r], lg[1][r]);
      mx = fmaxf(mx, __shfl_xor(mx, 1));
      mx = fmaxf(mx, __shfl_xor(mx, 2));
      mx = fmaxf(mx, __shfl_xor(mx, 4));
      mx = fmaxf(mx, __shfl_xor(mx, 8));
      const float mnew = fmaxf(mrow[r], mx);
      const float corr = __expf(mrow[r] - mnew);
      mrow[r] = mnew;
      const float e0 = __expf(lg[0][r] - mnew);
      const float e1 = __expf(lg[1][r] - mnew);
      p0[r] = e0; p1[r] = e1;
      float ts = e0 + e1;
      ts += __shfl_xor(ts, 1);
      ts += __shfl_xor(ts, 2);
      ts += __shfl_xor(ts, 4);
      ts += __shfl_xor(ts, 8);
      srow[r] = srow[r] * corr + ts;
      acc[0][r] *= corr; acc[1][r] *= corr; acc[2][r] *= corr; acc[3][r] *= corr;
    }

    // ---- transpose P (C-layout) -> A-layout through per-wave LDS tile
    bf16_t* pb = &Pst[wid][0];
    {
      const int row = (lane < 16) ? 0 : 8;
      for (int r = 0; r < 8; ++r) {
        pb[(row + r) * 32 + l16]      = (bf16_t)p0[r];
        pb[(row + r) * 32 + 16 + l16] = (bf16_t)p1[r];
      }
    }
    asm volatile("s_wait_dscnt 0" ::: "memory");   // cross-lane LDS RAW inside wave
    v16bf ap;
    {
      const bf16_t* p = pb + l16 * 32 + lhi * 8;
      ap = cat8(*(const v8bf*)p, *(const v8bf*)(p + 16));
    }

    // ---- O += P * V : 4 d-column tiles from V^T chunk in LDS
    for (int j = 0; j < 4; ++j) {
      const bf16_t* p = &Vs[cur][(j * 16 + l16) * 32 + lhi * 8];
      v16bf bb = cat8(*(const v8bf*)p, *(const v8bf*)(p + 16));
      acc[j] = wmma_bf16(ap, bb, acc[j]);
    }

    __syncthreads();                       // all waves done reading `cur`
    if (step + 1 < nsteps) publish();
  }

  // ---- normalize and write combined heads layout [B,S,HIDDEN] (bf16)
  const int rbase = (lane < 16) ? 0 : 8;
  for (int j = 0; j < 4; ++j)
    for (int r = 0; r < 8; ++r) {
      const int q = qbase + rbase + r;
      const int d = j * 16 + l16;
      const float o = acc[j][r] / srow[r];
      attn[((size_t)b * SEQ + q) * HIDDEN + h * HD + d] = (bf16_t)o;
    }
}

// ---------------------------------------------------------------- launch
extern "C" void kernel_launch(void* const* d_in, const int* in_sizes, int n_in,
                              void* d_out, int out_size, void* d_ws, size_t ws_size,
                              hipStream_t stream) {
  const float* query = (const float*)d_in[0];
  const float* masks = (const float*)d_in[1];
  const float* Wq    = (const float*)d_in[2];
  const float* bq    = (const float*)d_in[3];
  const float* Wk    = (const float*)d_in[4];
  const float* bk    = (const float*)d_in[5];
  const float* Wv    = (const float*)d_in[6];
  const float* bv    = (const float*)d_in[7];
  const float* Wo    = (const float*)d_in[8];
  const float* bo    = (const float*)d_in[9];

  const int Mrows = BATCH * SEQ;            // 8192
  const size_t MB = 1024ull * 1024ull;
  char* ws = (char*)d_ws;
  bf16_t* xbf  = (bf16_t*)(ws + 0);         // 16 MB  [M, C]
  bf16_t* wqb  = (bf16_t*)(ws + 16 * MB);   //  2 MB
  bf16_t* wkb  = (bf16_t*)(ws + 18 * MB);
  bf16_t* wvb  = (bf16_t*)(ws + 20 * MB);
  bf16_t* wob  = (bf16_t*)(ws + 22 * MB);
  bf16_t* Qhp  = (bf16_t*)(ws + 24 * MB);   // 16 MB  [B,H,S,D], pre-scaled
  bf16_t* Khp  = (bf16_t*)(ws + 40 * MB);   // 16 MB  [B,H,S,D]
  bf16_t* Vtp  = (bf16_t*)(ws + 56 * MB);   // 16 MB  [B,H,D,S]
  bf16_t* attn = (bf16_t*)(ws + 72 * MB);   // 16 MB  [B,S,C]

  const int nx = Mrows * HIDDEN;            // 8 Mi elems
  const int nw = HIDDEN * HIDDEN;           // 1 Mi elems
  f32_to_bf16_kernel<<<nx / 1024, 256, 0, stream>>>(query, xbf, nx);
  f32_to_bf16_kernel<<<nw / 1024, 256, 0, stream>>>(Wq, wqb, nw);
  f32_to_bf16_kernel<<<nw / 1024, 256, 0, stream>>>(Wk, wkb, nw);
  f32_to_bf16_kernel<<<nw / 1024, 256, 0, stream>>>(Wv, wvb, nw);
  f32_to_bf16_kernel<<<nw / 1024, 256, 0, stream>>>(Wo, wob, nw);

  dim3 ggrid(HIDDEN / BN, Mrows / BM);      // (8, 64)
  gemm_bf16_kernel<1><<<ggrid, 256, 0, stream>>>(xbf, wqb, bq, (void*)Qhp,
                                                 Mrows, HIDDEN, HIDDEN, 0.125f); // *D^-0.5
  gemm_bf16_kernel<1><<<ggrid, 256, 0, stream>>>(xbf, wkb, bk, (void*)Khp,
                                                 Mrows, HIDDEN, HIDDEN, 1.0f);
  gemm_bf16_kernel<2><<<ggrid, 256, 0, stream>>>(xbf, wvb, bv, (void*)Vtp,
                                                 Mrows, HIDDEN, HIDDEN, 1.0f);

  dim3 agrid(SEQ / 128, HEADS, BATCH);      // (16, 16, 4)
  attention_kernel<<<agrid, 256, 0, stream>>>(Qhp, Khp, Vtp, masks, attn);

  gemm_bf16_kernel<3><<<ggrid, 256, 0, stream>>>(attn, wob, bo, d_out,
                                                 Mrows, HIDDEN, HIDDEN, 1.0f);
}
